// Attention_80487687127167
// MI455X (gfx1250) — matrix operand order repaired and checked by reading, not compiled, not run on hardware
//
#include <hip/hip_runtime.h>
#include <hip/hip_bf16.h>

typedef __bf16 bf16_t;
typedef __attribute__((ext_vector_type(16))) __bf16 v16bf;
typedef __attribute__((ext_vector_type(8)))  float  v8f;

#define DIMC   1024
#define NHEAD  16
#define HDIM   64
#define SEQ    2048
#define BATCH  4
#define QKVC   (3*DIMC)

union FragU { v16bf v; uint4 q[2]; };
union Pack2 { bf16_t h[2]; unsigned int u; };

static __device__ __forceinline__ v8f wmma_bf16(v16bf a, v16bf b, v8f c) {
  return __builtin_amdgcn_wmma_f32_16x16x32_bf16(false, a, false, b, (short)0, c,
                                                 false, false);
}

// CDNA5 async copy global -> LDS (128-bit), tracked by ASYNCcnt.
static __device__ __forceinline__ void async_b128(unsigned lds_off,
                                                  const void* gaddr) {
  asm volatile("global_load_async_to_lds_b128 %0, %1, off"
               :: "v"(lds_off), "v"((unsigned long long)(size_t)gaddr)
               : "memory");
}
static __device__ __forceinline__ void wait_async0() {
  asm volatile("s_wait_asynccnt 0x0" ::: "memory");
}

// ---------------------------------------------------------------------------
// Kernel 0: W[k][n] f32  ->  Wt[n][k] bf16 (LDS-tiled 64x64 transpose+convert)
// ---------------------------------------------------------------------------
__global__ __launch_bounds__(256)
void wtr_kernel(const float* __restrict__ W, bf16_t* __restrict__ Wt,
                int K, int N) {
  __shared__ float lds[64][65];
  const int t = threadIdx.x;
  const int kBase = blockIdx.x * 64;
  const int nBase = blockIdx.y * 64;
  const int nl = t & 63, kc = t >> 6;
  #pragma unroll
  for (int r = 0; r < 16; ++r) {
    const int k = kc + r * 4;
    lds[k][nl] = W[(size_t)(kBase + k) * N + nBase + nl];
  }
  __syncthreads();
  const int kp = (t & 31) * 2, nc = t >> 5;
  #pragma unroll
  for (int r = 0; r < 8; ++r) {
    const int n = nc + r * 8;
    Pack2 p;
    p.h[0] = (bf16_t)lds[kp][n];
    p.h[1] = (bf16_t)lds[kp + 1][n];
    *(unsigned int*)(Wt + (size_t)(nBase + n) * K + kBase + kp) = p.u;
  }
}

// ---------------------------------------------------------------------------
// Kernel 1: qkv = x @ W_qkv + b_qkv. Wave = 32 rows x 64 cols. Q pre-scaled by
// D^-0.5*log2(e), layouts: Q,K -> [B,H,N,D]; V -> [B,H,D,N] (pre-transposed).
// ---------------------------------------------------------------------------
__global__ __launch_bounds__(128)
void qkv_kernel(const float* __restrict__ x, const bf16_t* __restrict__ Wt,
                const float* __restrict__ bqkv, bf16_t* __restrict__ Qb,
                bf16_t* __restrict__ Kb, bf16_t* __restrict__ Vt) {
  const int lane = threadIdx.x & 31;
  const int wave = threadIdx.x >> 5;
  const int hl = lane >> 4, ln = lane & 15;
  const int rowBase = blockIdx.x * 32;
  const int colBase = (blockIdx.y * 4 + wave) * 64;
  const float qscale = 0.125f * 1.44269504088896340736f;

  v8f acc[2][4] = {};
  const float* xr[2] = { x + (size_t)(rowBase + ln) * DIMC,
                         x + (size_t)(rowBase + 16 + ln) * DIMC };

  for (int k0 = 0; k0 < DIMC; k0 += 32) {
    v16bf af[2];
    #pragma unroll
    for (int rh = 0; rh < 2; ++rh) {
      const float* ap = xr[rh] + k0 + hl * 8;
      float4 a0 = *(const float4*)(ap);
      float4 a1 = *(const float4*)(ap + 4);
      float4 a2 = *(const float4*)(ap + 16);
      float4 a3 = *(const float4*)(ap + 20);
      af[rh][0]=(bf16_t)a0.x; af[rh][1]=(bf16_t)a0.y; af[rh][2]=(bf16_t)a0.z; af[rh][3]=(bf16_t)a0.w;
      af[rh][4]=(bf16_t)a1.x; af[rh][5]=(bf16_t)a1.y; af[rh][6]=(bf16_t)a1.z; af[rh][7]=(bf16_t)a1.w;
      af[rh][8]=(bf16_t)a2.x; af[rh][9]=(bf16_t)a2.y; af[rh][10]=(bf16_t)a2.z; af[rh][11]=(bf16_t)a2.w;
      af[rh][12]=(bf16_t)a3.x; af[rh][13]=(bf16_t)a3.y; af[rh][14]=(bf16_t)a3.z; af[rh][15]=(bf16_t)a3.w;
    }
    #pragma unroll
    for (int t = 0; t < 4; ++t) {
      const bf16_t* wrow = Wt + (size_t)(colBase + t * 16 + ln) * DIMC + k0 + hl * 8;
      FragU bfr;
      bfr.q[0] = *(const uint4*)(wrow);
      bfr.q[1] = *(const uint4*)(wrow + 16);
      acc[0][t] = wmma_bf16(af[0], bfr.v, acc[0][t]);
      acc[1][t] = wmma_bf16(af[1], bfr.v, acc[1][t]);
    }
  }

  // which is provably wave-uniform (colBase multiple of 64; tiles don't cross
  // a 1024 boundary) -> scalar branch, no EXEC-mask juggling.
  const int which = colBase >> 10;
  const int cb = colBase & 1023;

  if (which == 0) {                       // ---- Q: [B,H,N,D], scaled ----
    #pragma unroll
    for (int t = 0; t < 4; ++t) {
      const float bias = bqkv[colBase + t * 16 + ln];
      const int c2 = cb + t * 16 + ln;
      const int h = c2 >> 6, d = c2 & 63;
      #pragma unroll
      for (int rh = 0; rh < 2; ++rh)
        #pragma unroll
        for (int r = 0; r < 8; ++r) {
          const int row = rowBase + rh * 16 + r + hl * 8;
          const int b = row >> 11, n = row & (SEQ - 1);
          Qb[((size_t)((b*NHEAD + h)*SEQ + n))*HDIM + d] =
              (bf16_t)((acc[rh][t][r] + bias) * qscale);
        }
    }
  } else if (which == 1) {                // ---- K: [B,H,N,D] ----
    #pragma unroll
    for (int t = 0; t < 4; ++t) {
      const float bias = bqkv[colBase + t * 16 + ln];
      const int c2 = cb + t * 16 + ln;
      const int h = c2 >> 6, d = c2 & 63;
      #pragma unroll
      for (int rh = 0; rh < 2; ++rh)
        #pragma unroll
        for (int r = 0; r < 8; ++r) {
          const int row = rowBase + rh * 16 + r + hl * 8;
          const int b = row >> 11, n = row & (SEQ - 1);
          Kb[((size_t)((b*NHEAD + h)*SEQ + n))*HDIM + d] =
              (bf16_t)(acc[rh][t][r] + bias);
        }
    }
  } else {                                // ---- V: [B,H,D,N], packed b128 ----
    #pragma unroll
    for (int t = 0; t < 4; ++t) {
      const float bias = bqkv[colBase + t * 16 + ln];
      const int c2 = cb + t * 16 + ln;
      const int h = c2 >> 6, d = c2 & 63;
      #pragma unroll
      for (int rh = 0; rh < 2; ++rh) {
        const int row0 = rowBase + rh * 16 + hl * 8;   // 8 consecutive n
        const int b = row0 >> 11, n0 = row0 & (SEQ - 1);
        unsigned int w4[4];
        #pragma unroll
        for (int r = 0; r < 4; ++r) {
          Pack2 p;
          p.h[0] = (bf16_t)(acc[rh][t][2 * r] + bias);
          p.h[1] = (bf16_t)(acc[rh][t][2 * r + 1] + bias);
          w4[r] = p.u;
        }
        *(uint4*)(Vt + ((size_t)((b*NHEAD + h)*HDIM + d))*SEQ + n0) =
            make_uint4(w4[0], w4[1], w4[2], w4[3]);
      }
    }
  }
}

// ---------------------------------------------------------------------------
// Kernel 2: flash attention. 4 waves/block share one (b,h); K/V tiles staged
// into double-buffered LDS with async global->LDS copies. One wave per
// 16-query tile: S^T = K Q^T, online softmax, O^T += V^T P^T.
// ---------------------------------------------------------------------------
__global__ __launch_bounds__(128)
void attn_kernel(const bf16_t* __restrict__ Qb, const bf16_t* __restrict__ Kb,
                 const bf16_t* __restrict__ Vt, bf16_t* __restrict__ Ob) {
  __shared__ bf16_t Ks[2][32][64 + 8];   // +16B pad: bank spread, keeps 16B align
  __shared__ bf16_t Vs[2][64][32 + 8];
  const int tid  = threadIdx.x;
  const int lane = tid & 31;
  const int wave = tid >> 5;
  const int hl = lane >> 4, ln = lane & 15;
  const int bh = blockIdx.x >> 5;              // 64 (b,h) pairs
  const int qt = (blockIdx.x & 31) * 4 + wave; // 128 q-tiles per head
  const int b = bh >> 4, h = bh & 15;
  const size_t base = (size_t)bh * SEQ * HDIM; // same for K ([N,D]) and Vt ([D,N])
  const int qbase = qt * 16;

  // Q as B-fragments (lane = query): frag f element i -> d = f*32 + 8*hl + (i&7) + 16*(i>>3)
  FragU qf[2];
  {
    const bf16_t* qrow = Qb + base + (size_t)(qbase + ln) * HDIM;
    qf[0].q[0] = *(const uint4*)(qrow + hl * 8);
    qf[0].q[1] = *(const uint4*)(qrow + 16 + hl * 8);
    qf[1].q[0] = *(const uint4*)(qrow + 32 + hl * 8);
    qf[1].q[1] = *(const uint4*)(qrow + 48 + hl * 8);
  }

  auto stage = [&](int kb, int buf) {
    // K tile: 32 rows x 128B -> 256 x b128 chunks, 2 per thread
    #pragma unroll
    for (int j = 0; j < 2; ++j) {
      const int c = tid * 2 + j;
      const int row = c >> 3, sub = c & 7;
      async_b128((unsigned)(size_t)&Ks[buf][row][sub * 8],
                 Kb + base + (size_t)(kb + row) * HDIM + sub * 8);
    }
    // V tile (d-major): 64 rows x 64B -> 256 x b128 chunks, 2 per thread
    #pragma unroll
    for (int j = 0; j < 2; ++j) {
      const int c = tid * 2 + j;
      const int row = c >> 2, sub = c & 3;
      async_b128((unsigned)(size_t)&Vs[buf][row][sub * 8],
                 Vt + base + (size_t)row * SEQ + kb + sub * 8);
    }
  };

  stage(0, 0);
  wait_async0();
  __syncthreads();

  float mrun = -__builtin_inff();
  float lrun = 0.0f;
  v8f oacc[4] = {};

  for (int kb = 0; kb < SEQ; kb += 32) {
    const int cur = (kb >> 5) & 1;
    if (kb + 32 < SEQ) stage(kb + 32, cur ^ 1);  // prefetch next tile

    // ---- S^T for two 16-key tiles (fragments from LDS) ----
    v8f s[2];
    #pragma unroll
    for (int t = 0; t < 2; ++t) {
      const bf16_t* krow = &Ks[cur][t * 16 + ln][0];
      FragU k0, k1;
      k0.q[0] = *(const uint4*)(krow + hl * 8);
      k0.q[1] = *(const uint4*)(krow + 16 + hl * 8);
      k1.q[0] = *(const uint4*)(krow + 32 + hl * 8);
      k1.q[1] = *(const uint4*)(krow + 48 + hl * 8);
      v8f st = {};
      st = wmma_bf16(k0.v, qf[0].v, st);
      st = wmma_bf16(k1.v, qf[1].v, st);
      s[t] = st;
    }

    // ---- online softmax (query = ln; keys split across lane halves) ----
    float mloc = s[0][0];
    #pragma unroll
    for (int r = 0; r < 8; ++r) {
      mloc = fmaxf(mloc, s[0][r]);
      mloc = fmaxf(mloc, s[1][r]);
    }
    mloc = fmaxf(mloc, __shfl_xor(mloc, 16, 32));
    const float mnew = fmaxf(mrun, mloc);
    const float corr = exp2f(mrun - mnew);
    float p0[8], p1[8];
    float rs = 0.0f;
    #pragma unroll
    for (int r = 0; r < 8; ++r) {
      p0[r] = exp2f(s[0][r] - mnew);
      p1[r] = exp2f(s[1][r] - mnew);
      rs += p0[r] + p1[r];
    }
    rs += __shfl_xor(rs, 16, 32);
    lrun = lrun * corr + rs;
    mrun = mnew;
    #pragma unroll
    for (int c = 0; c < 4; ++c)
      #pragma unroll
      for (int r = 0; r < 8; ++r) oacc[c][r] *= corr;

    // ---- P^T B-fragment (element i -> key kb + i + 16*hl) ----
    v16bf pf;
    #pragma unroll
    for (int r = 0; r < 8; ++r) {
      const float u = hl ? p0[r] : p1[r];
      const float w = __shfl_xor(u, 16, 32);
      const float e0 = hl ? w : p0[r];
      const float e1 = hl ? p1[r] : w;
      pf[r]     = (bf16_t)e0;
      pf[8 + r] = (bf16_t)e1;
    }

    // ---- O^T += V^T P^T : 4 d-chunks of 16 rows (V^T rows from LDS) ----
    #pragma unroll
    for (int c = 0; c < 4; ++c) {
      const bf16_t* vrow = &Vs[cur][c * 16 + ln][0];
      FragU vf;
      vf.q[0] = *(const uint4*)(vrow + hl * 16);
      vf.q[1] = *(const uint4*)(vrow + hl * 16 + 8);
      oacc[c] = wmma_bf16(vf.v, pf, oacc[c]);
    }

    wait_async0();     // my prefetch done
    __syncthreads();   // everyone staged next + done reading cur
  }

  // ---- normalize and write O[b, qbase+ln, h*64 + d] ----
  const float rl = 1.0f / lrun;
  bf16_t* orow = Ob + ((size_t)(b * SEQ + qbase + ln)) * DIMC + h * HDIM;
  #pragma unroll
  for (int c = 0; c < 4; ++c) {
    unsigned int w4[4];
    #pragma unroll
    for (int r = 0; r < 4; ++r) {
      Pack2 p;
      p.h[0] = (bf16_t)(oacc[c][2 * r] * rl);
      p.h[1] = (bf16_t)(oacc[c][2 * r + 1] * rl);
      w4[r] = p.u;
    }
    *(uint4*)(orow + c * 16 + hl * 8) = make_uint4(w4[0], w4[1], w4[2], w4[3]);
  }
}

// ---------------------------------------------------------------------------
// Kernel 3: out = O @ W_proj + b_proj (f32 out). Wave = 32 rows x 64 cols,
// both operands bf16 with packed 128-bit loads.
// ---------------------------------------------------------------------------
__global__ __launch_bounds__(128)
void proj_kernel(const bf16_t* __restrict__ Ob, const bf16_t* __restrict__ Wt,
                 const float* __restrict__ bp, float* __restrict__ out) {
  const int lane = threadIdx.x & 31;
  const int wave = threadIdx.x >> 5;
  const int hl = lane >> 4, ln = lane & 15;
  const int rowBase = blockIdx.x * 32;
  const int colBase = (blockIdx.y * 4 + wave) * 64;

  v8f acc[2][4] = {};
  const bf16_t* ar[2] = { Ob + (size_t)(rowBase + ln) * DIMC,
                          Ob + (size_t)(rowBase + 16 + ln) * DIMC };

  for (int k0 = 0; k0 < DIMC; k0 += 32) {
    FragU af[2];
    #pragma unroll
    for (int rh = 0; rh < 2; ++rh) {
      af[rh].q[0] = *(const uint4*)(ar[rh] + k0 + hl * 8);
      af[rh].q[1] = *(const uint4*)(ar[rh] + k0 + 16 + hl * 8);
    }
    #pragma unroll
    for (int t = 0; t < 4; ++t) {
      const bf16_t* wrow = Wt + (size_t)(colBase + t * 16 + ln) * DIMC + k0 + hl * 8;
      FragU bfr;
      bfr.q[0] = *(const uint4*)(wrow);
      bfr.q[1] = *(const uint4*)(wrow + 16);
      acc[0][t] = wmma_bf16(af[0].v, bfr.v, acc[0][t]);
      acc[1][t] = wmma_bf16(af[1].v, bfr.v, acc[1][t]);
    }
  }

  #pragma unroll
  for (int t = 0; t < 4; ++t) {
    const int col = colBase + t * 16 + ln;
    const float bias = bp[col];
    #pragma unroll
    for (int rh = 0; rh < 2; ++rh)
      #pragma unroll
      for (int r = 0; r < 8; ++r)
        out[(size_t)(rowBase + rh * 16 + r + hl * 8) * DIMC + col] =
            acc[rh][t][r] + bias;
  }
}

// ---------------------------------------------------------------------------
extern "C" void kernel_launch(void* const* d_in, const int* in_sizes, int n_in,
                              void* d_out, int out_size, void* d_ws, size_t ws_size,
                              hipStream_t stream) {
  const float* x    = (const float*)d_in[0];
  const float* Wqkv = (const float*)d_in[1];
  const float* bqkv = (const float*)d_in[2];
  const float* Wp   = (const float*)d_in[3];
  const float* bp   = (const float*)d_in[4];
  float* out = (float*)d_out;

  const size_t nQ = (size_t)BATCH * NHEAD * SEQ * HDIM;  // 8Mi elems each
  bf16_t* Qb  = (bf16_t*)d_ws;
  bf16_t* Kb  = Qb + nQ;
  bf16_t* Vt  = Kb + nQ;
  bf16_t* Ob  = Vt + nQ;
  bf16_t* Wqt = Ob + nQ;                   // [3072][1024] bf16
  bf16_t* Wpt = Wqt + (size_t)QKVC * DIMC; // [1024][1024] bf16

  dim3 blk(128);
  // weight transpose+convert: tiles of 64x64
  wtr_kernel<<<dim3(DIMC / 64, QKVC / 64), 256, 0, stream>>>(Wqkv, Wqt, DIMC, QKVC);
  wtr_kernel<<<dim3(DIMC / 64, DIMC / 64), 256, 0, stream>>>(Wp, Wpt, DIMC, DIMC);
  // 8192 rows -> 256 tiles of 32; 3072 cols -> 12 * (4 waves * 64)
  qkv_kernel<<<dim3(256, 12), blk, 0, stream>>>(x, Wqt, bqkv, Qb, Kb, Vt);
  // 64 (b,h) x 32 blocks of 4 q-tiles
  attn_kernel<<<dim3(2048), blk, 0, stream>>>(Qb, Kb, Vt, Ob);
  // proj: 1024 cols -> 4 * (4 waves * 64)
  proj_kernel<<<dim3(256, 4), blk, 0, stream>>>(Ob, Wpt, bp, out);
}
